// UHGMultiHeadAttention_45981919871207
// MI455X (gfx1250) — matrix operand order repaired and checked by reading, not compiled, not run on hardware
//
#include <hip/hip_runtime.h>
#include <math.h>

#define EPS 1e-6f
#define Bn 4
#define Sn 512
#define FDn 64
#define Hn 4
#define Dn 65

typedef __attribute__((ext_vector_type(16))) _Float16 v16h;
typedef __attribute__((ext_vector_type(8)))  _Float16 v8h;
typedef __attribute__((ext_vector_type(8)))  float    v8f;

// ---------------- workspace layout (floats) ----------------
#define WEFF_OFF 0                         // 13 matrices of 65x65 (q0..3,k0..3,v0..3,Wo)
#define WEFF_SZ  (13 * 4225)
#define XT_OFF   54928                     // transformed q/k/v: [m=proj*4+h][token(2048)][65]
#define XT_SZ    (12 * 2048 * 65)
#define AGG_OFF  (XT_OFF + XT_SZ)          // aggregate results: [h][token][65]
#define AGG_SZ   (4 * 2048 * 65)
#define WT_OFF   (AGG_OFF + AGG_SZ)        // transposed weights: [h][b][j][i]
#define WT_SZ    (4 * 4 * 512 * 512)
#define WS_FULL_FLOATS ((size_t)WT_OFF + (size_t)WT_SZ)

// ============================================================
// Kernel 1: orthogonalize via Newton-Schulz polar iteration.
// u @ vt from SVD == orthogonal polar factor; NS converges to it.
// One block per matrix (12x 64x64 blocks of W_q/k/v, 1x 65x65 W_o).
// ============================================================
__global__ __launch_bounds__(256) void ortho_kernel(
    const float* __restrict__ Wq, const float* __restrict__ Wk,
    const float* __restrict__ Wv, const float* __restrict__ Wo,
    float* __restrict__ wEff) {
  __shared__ float X[66 * 66];
  __shared__ float T[66 * 66];
  __shared__ float red[256];
  const int m = blockIdx.x;          // 0..12
  const int N = (m < 12) ? 64 : 65;
  const int tid = threadIdx.x;
  const float* src;
  if (m < 12) {
    int proj = m >> 2, h = m & 3;
    const float* base = (proj == 0) ? Wq : (proj == 1 ? Wk : Wv);
    src = base + (size_t)h * 4225;
  } else {
    src = Wo;
  }
  // load N x N block (row-major stride 65 in source)
  for (int idx = tid; idx < N * N; idx += 256) {
    int r = idx / N, c = idx % N;
    X[r * 66 + c] = src[r * 65 + c];
  }
  __syncthreads();
  // Frobenius norm
  float psum = 0.f;
  for (int idx = tid; idx < N * N; idx += 256) {
    int r = idx / N, c = idx % N;
    float v = X[r * 66 + c];
    psum += v * v;
  }
  red[tid] = psum;
  __syncthreads();
  for (int s = 128; s > 0; s >>= 1) {
    if (tid < s) red[tid] += red[tid + s];
    __syncthreads();
  }
  const float inv = rsqrtf(red[0] + 1e-30f);
  for (int idx = tid; idx < N * N; idx += 256) {
    int r = idx / N, c = idx % N;
    X[r * 66 + c] *= inv;
  }
  __syncthreads();
  // Newton-Schulz: X <- 1.5 X - 0.5 X (X^T X)
  for (int it = 0; it < 30; it++) {
    for (int idx = tid; idx < N * N; idx += 256) {
      int r = idx / N, c = idx % N;
      float s = 0.f;
      for (int k = 0; k < N; k++) s += X[k * 66 + r] * X[k * 66 + c];
      T[r * 66 + c] = s;
    }
    __syncthreads();
    float y[17];
    int cnt = 0;
    for (int idx = tid; idx < N * N; idx += 256) {
      int r = idx / N, c = idx % N;
      float s = 0.f;
      for (int k = 0; k < N; k++) s += X[r * 66 + k] * T[k * 66 + c];
      y[cnt++] = 1.5f * X[r * 66 + c] - 0.5f * s;
    }
    __syncthreads();
    cnt = 0;
    for (int idx = tid; idx < N * N; idx += 256) {
      int r = idx / N, c = idx % N;
      X[r * 66 + c] = y[cnt++];
    }
    __syncthreads();
  }
  // write effective 65x65 W: identity with top-left NxN = polar factor
  for (int idx = tid; idx < 4225; idx += 256) {
    int r = idx / 65, c = idx % 65;
    float v = (r < N && c < N) ? X[r * 66 + c] : (r == c ? 1.f : 0.f);
    wEff[(size_t)m * 4225 + idx] = v;
  }
}

// ============================================================
// Kernel 2: projective transform of q/k/v via WMMA f16->f32.
// t_pre = [x|1] @ W^T, tiled 16(M) x 80(N) x 96(K), zero-padded.
// Both WMMA operands stored so each lane's 16-elem fragment is
// two runs of 8 contiguous halves, 16 apart, in LDS -> two aligned ds_load_b128 each.
// grid = (32, 12): blockIdx.x = 64-token chunk, blockIdx.y = proj*4+h
// ============================================================
__global__ __launch_bounds__(128) void transform_kernel(
    const float* __restrict__ q, const float* __restrict__ k,
    const float* __restrict__ v, const float* __restrict__ wEff,
    float* __restrict__ xt) {
  __shared__ _Float16 BsT[80 * 96];  // BsT[n][k] = W[n][k], row-major, padded
  __shared__ _Float16 As[64 * 96];   // As[tok][k] = Xaug[tok][k]
  __shared__ float    Ts[64 * 80];   // t_pre results
  __shared__ float    Sc[64];        // per-token scale
  const int tid = threadIdx.x;
  const int m = blockIdx.y;
  const int proj = m >> 2;
  const float* x = (proj == 0) ? q : (proj == 1 ? k : v);
  const float* W = wEff + (size_t)m * 4225;
  for (int idx = tid; idx < 80 * 96; idx += 128) {
    int nn = idx / 96, kk = idx % 96;
    float val = (kk < 65 && nn < 65) ? W[nn * 65 + kk] : 0.f;
    BsT[idx] = (_Float16)val;
  }
  const int tokBase = blockIdx.x * 64;
  for (int idx = tid; idx < 64 * 96; idx += 128) {
    int tl = idx / 96, kk = idx % 96;
    float val = (kk < 64) ? x[(size_t)(tokBase + tl) * 64 + kk]
                          : (kk == 64 ? 1.f : 0.f);
    As[idx] = (_Float16)val;
  }
  __syncthreads();
  const int wave = tid >> 5, lane = tid & 31;
  const int mrow = lane & 15, khalf = lane >> 4;
  const _Float16* arow = &As[(wave * 16 + mrow) * 96];
  for (int nt = 0; nt < 5; nt++) {
    v8f acc = {};
    const _Float16* brow = &BsT[(nt * 16 + mrow) * 96];
    #pragma unroll
    for (int kt = 0; kt < 3; kt++) {
      // A 16x32 f16 layout: elems 0-7 at K = kt*32+khalf*8+(0..7),
      //                     elems 8-15 at K = kt*32+16+khalf*8+(0..7)
      const _Float16* ap = arow + kt * 32 + khalf * 8;
      v8h aLo = *(const v8h*)(ap);
      v8h aHi = *(const v8h*)(ap + 16);
      v16h a = __builtin_shufflevector(aLo, aHi, 0, 1, 2, 3, 4, 5, 6, 7,
                                       8, 9, 10, 11, 12, 13, 14, 15);
      // B 32x16 f16 layout: lane = N col, elems 0-7 at K = kt*32+khalf*8+(0..7),
      //                     elems 8-15 at K = kt*32+16+khalf*8+(0..7)
      const _Float16* bp = brow + kt * 32 + khalf * 8;
      v8h bLo = *(const v8h*)(bp);
      v8h bHi = *(const v8h*)(bp + 16);
      v16h bvec = __builtin_shufflevector(bLo, bHi, 0, 1, 2, 3, 4, 5, 6, 7,
                                          8, 9, 10, 11, 12, 13, 14, 15);
      acc = __builtin_amdgcn_wmma_f32_16x16x32_f16(
          false, a, false, bvec, (short)0, acc, false, false);
    }
    // D layout: VGPR r -> row khalf*8+r, col = lane&15
    #pragma unroll
    for (int r = 0; r < 8; r++)
      Ts[(wave * 16 + khalf * 8 + r) * 80 + nt * 16 + mrow] = acc[r];
  }
  __syncthreads();
  if (tid < 64) {
    float t64 = Ts[tid * 80 + 64];
    float t1  = Ts[tid * 80 + 1];
    float t2  = Ts[tid * 80 + 2];
    float s1 = 1.f / (t64 + EPS);
    float aa = t1 * s1, bb = t2 * s1, cc = t64 * s1;
    float cr = fabsf(aa - cc) / (fabsf(bb - cc) + EPS);
    float s2 = 1.f / (cr + EPS);
    float s3 = 1.f / (cc * s2 + EPS);
    Sc[tid] = s1 * s2 * s3;
  }
  __syncthreads();
  for (int idx = tid; idx < 64 * 65; idx += 128) {
    int tl = idx / 65, dd = idx % 65;
    xt[((size_t)m * 2048 + tokBase + tl) * 65 + dd] = Ts[tl * 80 + dd] * Sc[tl];
  }
}

// ============================================================
// Kernel 3: attention scores + softmax.
// One block per (b,h,i) row; writes weights output (B,H,S,S) and
// optionally the (h,b,j,i)-transposed copy for coalesced aggregate reads.
// ============================================================
__global__ __launch_bounds__(256) void scores_kernel(
    const float* __restrict__ xt, float* __restrict__ wOut,
    float* __restrict__ wT) {
  __shared__ float sbuf[512];
  __shared__ float red[256];
  const int tid = threadIdx.x;
  const int r = blockIdx.x;              // 0..8191
  const int b = r / (Hn * Sn);
  const int rem = r % (Hn * Sn);
  const int h = rem / Sn;
  const int i = rem % Sn;
  const size_t qb = ((size_t)h * 2048 + b * 512 + i) * 65;
  const float q0 = xt[qb + 0];
  const float q1 = xt[qb + 1];
  float lmax = -1e30f;
  for (int j = tid; j < 512; j += 256) {
    const float* kr = xt + ((size_t)(4 + h) * 2048 + b * 512 + j) * 65;
    float k0 = kr[0], k1 = kr[1];
    float cr = fabsf(q0 * k1) / (fabsf(q1 * k0) + EPS);
    float s = 1.f / (1.f + fabsf(logf(cr + EPS)));
    sbuf[j] = s;
    lmax = fmaxf(lmax, s);
  }
  red[tid] = lmax;
  __syncthreads();
  for (int s = 128; s > 0; s >>= 1) {
    if (tid < s) red[tid] = fmaxf(red[tid], red[tid + s]);
    __syncthreads();
  }
  const float mx = red[0];
  __syncthreads();
  float lsum = 0.f;
  for (int j = tid; j < 512; j += 256) {
    float e = expf(sbuf[j] - mx);
    sbuf[j] = e;
    lsum += e;
  }
  red[tid] = lsum;
  __syncthreads();
  for (int s = 128; s > 0; s >>= 1) {
    if (tid < s) red[tid] += red[tid + s];
    __syncthreads();
  }
  const float rinv = 1.f / red[0];
  for (int j = tid; j < 512; j += 256) {
    float wv = sbuf[j] * rinv;
    wOut[(((size_t)b * Hn + h) * Sn + i) * Sn + j] = wv;
    if (wT) wT[(((size_t)h * Bn + b) * Sn + j) * Sn + i] = wv;
  }
}

// ============================================================
// Kernel 4: sequential projective aggregate scan.
// One lane per (h,b,i) row; point/line/v held in registers,
// D=65 loops fully unrolled (rolls become static indices).
// v_j is workgroup-uniform -> scalar-load friendly.
// ============================================================
__global__ __launch_bounds__(256) void aggregate_kernel(
    const float* __restrict__ xt, const float* __restrict__ wT,
    const float* __restrict__ wOut, float* __restrict__ agg) {
  const int blk = blockIdx.x;            // 32 blocks: h(4) x b(4) x chunk(2)
  const int h = blk >> 3;
  const int b = (blk >> 1) & 3;
  const int chunk = blk & 1;
  const int i = chunk * 256 + threadIdx.x;
  float pt[65];
  #pragma unroll
  for (int d = 0; d < 65; d++) pt[d] = 0.f;
  pt[64] = 1.f;
  for (int j = 0; j < 512; j++) {
    const float* vr = xt + ((size_t)(8 + h) * 2048 + b * 512 + j) * 65;
    float vb[65];
    #pragma unroll
    for (int d = 0; d < 65; d++) vb[d] = vr[d];
    float w = wT ? wT[(((size_t)h * Bn + b) * Sn + j) * Sn + i]
                 : wOut[(((size_t)b * Hn + h) * Sn + i) * Sn + j];
    float line[65];
    #pragma unroll
    for (int d = 0; d < 65; d++) {
      int r1 = (d + 1 >= 65) ? d + 1 - 65 : d + 1;
      int r2 = (d + 2 >= 65) ? d + 2 - 65 : d + 2;
      line[d] = pt[r1] * vb[r2] - pt[r2] * vb[r1];
    }
    // p = join(line, w*vb) = w * join(line, vb); p[64] uses rolls (0,1)
    float q64 = line[0] * vb[1] - line[1] * vb[0];
    float scale = w / (w * q64 + EPS);
    #pragma unroll
    for (int d = 0; d < 65; d++) {
      int r1 = (d + 1 >= 65) ? d + 1 - 65 : d + 1;
      int r2 = (d + 2 >= 65) ? d + 2 - 65 : d + 2;
      pt[d] = (line[r1] * vb[r2] - line[r2] * vb[r1]) * scale;
    }
  }
  float* dst = agg + ((size_t)h * 2048 + b * 512 + i) * 65;
  #pragma unroll
  for (int d = 0; d < 65; d++) dst[d] = pt[d];
}

// ============================================================
// Kernel 5: final W_o transform + epilogue.
// out = concat(heads) (B,S,260); only t[0:65] (= head0_agg @ Wo^T) and
// t[259] (= head3_agg[64]) matter for out[..., :64].
// ============================================================
__global__ __launch_bounds__(256) void final_kernel(
    const float* __restrict__ agg, const float* __restrict__ wEff,
    float* __restrict__ out) {
  __shared__ float Wo[4225];
  const int tid = threadIdx.x;
  for (int idx = tid; idx < 4225; idx += 256)
    Wo[idx] = wEff[(size_t)12 * 4225 + idx];
  __syncthreads();
  const int token = blockIdx.x * 256 + tid;   // 0..2047
  float a0[65];
  const float* src = agg + (size_t)token * 65;          // head 0
  #pragma unroll
  for (int c = 0; c < 65; c++) a0[c] = src[c];
  const float z = agg[((size_t)3 * 2048 + token) * 65 + 64]; // head 3, last comp
  float y[64];
  for (int r = 0; r < 64; r++) {
    float s = 0.f;
    #pragma unroll
    for (int c = 0; c < 65; c++) s += a0[c] * Wo[r * 65 + c];
    y[r] = s;
  }
  float s1 = 1.f / (z + EPS);
  float aa = y[1] * s1, bb = y[2] * s1, cc = z * s1;
  float cr = fabsf(aa - cc) / (fabsf(bb - cc) + EPS);
  float s2 = 1.f / (cr + EPS);
  float s3 = 1.f / (cc * s2 + EPS);
  float scale = s1 * s2 * s3;
  float* dst = out + (size_t)token * 64;
  #pragma unroll
  for (int r = 0; r < 64; r++) dst[r] = y[r] * scale;
}

// ============================================================
extern "C" void kernel_launch(void* const* d_in, const int* in_sizes, int n_in,
                              void* d_out, int out_size, void* d_ws,
                              size_t ws_size, hipStream_t stream) {
  (void)in_sizes; (void)n_in; (void)out_size;
  const float* query = (const float*)d_in[0];
  const float* key   = (const float*)d_in[1];
  const float* value = (const float*)d_in[2];
  const float* Wq    = (const float*)d_in[3];
  const float* Wk    = (const float*)d_in[4];
  const float* Wv    = (const float*)d_in[5];
  const float* Wo    = (const float*)d_in[6];
  float* out  = (float*)d_out;                       // (B,S,64)
  float* wOut = out + (size_t)Bn * Sn * FDn;         // (B,H,S,S)
  float* wsf  = (float*)d_ws;
  float* wEff = wsf + WEFF_OFF;
  float* xt   = wsf + XT_OFF;
  float* agg  = wsf + AGG_OFF;
  const bool useWT = ws_size >= WS_FULL_FLOATS * sizeof(float);
  float* wT = useWT ? (wsf + WT_OFF) : nullptr;

  ortho_kernel<<<13, 256, 0, stream>>>(Wq, Wk, Wv, Wo, wEff);
  transform_kernel<<<dim3(32, 12), 128, 0, stream>>>(query, key, value, wEff, xt);
  scores_kernel<<<Bn * Hn * Sn, 256, 0, stream>>>(xt, wOut, wT);
  aggregate_kernel<<<32, 256, 0, stream>>>(xt, wT, wOut, agg);
  final_kernel<<<Sn * Bn / 256, 256, 0, stream>>>(agg, wEff, out);
}
